// ViT_9998683865339
// MI455X (gfx1250) — matrix operand order repaired and checked by reading, not compiled, not run on hardware
//
#include <hip/hip_runtime.h>
#include <hip/hip_bf16.h>

// ---------------- config ----------------
#define BATCH   32
#define IMG     224
#define NPAT    14
#define PSZ     16
#define PDIM    256        // 16*16
#define DMODEL  768
#define NHEADS  12
#define LAYERS  12
#define DQ      64
#define SEQ     197        // 14*14 + 1
#define SEQP    200        // padded score row stride (multiple of 4 -> b128 alignment)
#define MLPH    3072
#define NCLS    1000

typedef __attribute__((ext_vector_type(16))) _Float16 v16h;
typedef __attribute__((ext_vector_type(8)))  _Float16 v8h;
typedef __attribute__((ext_vector_type(4)))  _Float16 v4h;
typedef __attribute__((ext_vector_type(8)))  float    v8f;

#define LDS_STRIDE 40   // halves per row: 32 data + 8 pad; keeps 16B alignment

// =====================================================================
// Generic batched GEMM: C[z] = alpha * A[z] (M,K) * B[z] (K,N) [+bias][gelu][+res]
// fp32 inputs converted to f16 in LDS; f32 accumulation via v_wmma_f32_16x16x32_f16.
// Block: 256 threads (8 waves), 128xBN tile (BN = 128 or 64). Wave grid 4x2:
// each wave computes 32 x (BN/2) via 2 x (BN/32) WMMA frags -> 8 (or 4)
// back-to-back WMMAs per K-step. Double-buffered LDS, software pipelined:
// next tile's b128 global loads are issued before the WMMAs, f16 convert +
// LDS store after. Interior tiles are fully branchless; edges use clamped
// scalar loads (uniform scalar branch selects path).
// REQUIRES: lda % 4 == 0, ldb % 4 == 0, A/B 16B-aligned (all call sites comply).
// =====================================================================
template<int BN>
__global__ __launch_bounds__(256) void gemm_wmma_kernel(
    const float* __restrict__ A,  int lda,  long long sAz,
    const float* __restrict__ Bm, int ldb,  long long sBz, int transB,
    float* __restrict__ C,        int ldc,  long long sCz,
    const float* __restrict__ bias, long long sBiasz,
    const float* __restrict__ Res,  int ldres,
    int M, int N, int K, float alpha, int gelu)
{
    constexpr int BM = 128;
    constexpr int NF = BN / 32;       // B frags per wave (wave covers BN/2 cols)
    constexpr int CG = BN / 4;        // column groups in B tile

    __shared__ alignas(16) _Float16 As[2][BM * LDS_STRIDE];
    __shared__ alignas(16) _Float16 Bs[2][BN * LDS_STRIDE];   // stored [col][k]

    const int z = blockIdx.z;
    A  += (size_t)z * (size_t)sAz;
    Bm += (size_t)z * (size_t)sBz;
    C  += (size_t)z * (size_t)sCz;
    if (bias) bias += (size_t)z * (size_t)sBiasz;

    const int tid  = threadIdx.x;
    const int lane = tid & 31;
    const int wave = tid >> 5;
    const int wm = wave >> 1;          // 0..3 (rows)
    const int wn = wave & 1;           // 0..1 (cols)
    const int tm = blockIdx.y * BM;
    const int tn = blockIdx.x * BN;
    const int hi = lane >> 4;          // 0/1
    const int lo = lane & 15;

    // Register staging. Layout:
    //   rA[i*4+j] = A[tm + ((tid+i*256)>>3)][k0 + ((tid+i*256)&7)*4 + j]   (i<4, j<4)
    //   rB[j*4+c] = B[k0 + bK0 + j][tn + bC0 + c]                         (j<4, c<4)
    float rA[16];
    float rB[16];
    const int  bC0  = (tid % CG) * 4;            // B col group
    const int  bK0  = ((tid / CG) & 7) * 4;      // B k group
    const bool bAct = tid < CG * 8;              // threads participating in B stage

    auto loadA_clean = [&](int k0) {
        #pragma unroll
        for (int i = 0; i < 4; ++i) {
            int q  = tid + i * 256;            // 0..1023
            int r  = q >> 3;
            int kq = (q & 7) * 4;
            const float4 v = *(const float4*)(A + (size_t)(tm + r) * lda + (k0 + kq));
            rA[i*4+0] = v.x; rA[i*4+1] = v.y; rA[i*4+2] = v.z; rA[i*4+3] = v.w;
        }
    };
    auto loadA_guard = [&](int k0) {
        #pragma unroll
        for (int i = 0; i < 4; ++i) {
            int q  = tid + i * 256;
            int r  = q >> 3;
            int kq = (q & 7) * 4;
            int gr = tm + r;
            bool okr = gr < M;
            int cr = okr ? gr : 0;
            #pragma unroll
            for (int j = 0; j < 4; ++j) {
                int gk = k0 + kq + j;
                bool ok = okr && (gk < K);
                int ck = (gk < K) ? gk : 0;
                float v = A[(size_t)cr * lda + ck];
                rA[i*4+j] = ok ? v : 0.0f;
            }
        }
    };
    auto loadB_clean = [&](int k0) {
        if (!bAct) return;
        #pragma unroll
        for (int j = 0; j < 4; ++j) {
            const float4 v = *(const float4*)(Bm + (size_t)(k0 + bK0 + j) * ldb + tn + bC0);
            rB[j*4+0] = v.x; rB[j*4+1] = v.y; rB[j*4+2] = v.z; rB[j*4+3] = v.w;
        }
    };
    auto loadB_clean_t = [&](int k0) {
        if (!bAct) return;
        #pragma unroll
        for (int c = 0; c < 4; ++c) {
            const float4 v = *(const float4*)(Bm + (size_t)(tn + bC0 + c) * ldb + k0 + bK0);
            rB[0*4+c] = v.x; rB[1*4+c] = v.y; rB[2*4+c] = v.z; rB[3*4+c] = v.w;
        }
    };
    auto loadB_guard = [&](int k0) {
        if (!bAct) return;
        #pragma unroll
        for (int j = 0; j < 4; ++j) {
            int gk = k0 + bK0 + j;
            bool okk = gk < K;
            int ck = okk ? gk : 0;
            #pragma unroll
            for (int c = 0; c < 4; ++c) {
                int gc = tn + bC0 + c;
                bool ok = okk && (gc < N);
                int cc = (gc < N) ? gc : 0;
                float v = transB ? Bm[(size_t)cc * ldb + ck]
                                 : Bm[(size_t)ck * ldb + cc];
                rB[j*4+c] = ok ? v : 0.0f;
            }
        }
    };
    auto loadTile = [&](int k0) {
        bool fullK = (k0 + 32) <= K;                   // uniform
        if ((tm + BM) <= M && fullK) loadA_clean(k0); else loadA_guard(k0);
        if ((tn + BN) <= N && fullK) { if (transB) loadB_clean_t(k0); else loadB_clean(k0); }
        else loadB_guard(k0);
    };
    auto storeTile = [&](int buf) {
        #pragma unroll
        for (int i = 0; i < 4; ++i) {
            int q  = tid + i * 256;
            int r  = q >> 3;
            int kq = (q & 7) * 4;
            v4h h = { (_Float16)rA[i*4+0], (_Float16)rA[i*4+1],
                      (_Float16)rA[i*4+2], (_Float16)rA[i*4+3] };
            *(v4h*)(&As[buf][r * LDS_STRIDE + kq]) = h;
        }
        if (bAct) {
            #pragma unroll
            for (int c = 0; c < 4; ++c) {
                v4h h = { (_Float16)rB[0*4+c], (_Float16)rB[1*4+c],
                          (_Float16)rB[2*4+c], (_Float16)rB[3*4+c] };
                *(v4h*)(&Bs[buf][(bC0 + c) * LDS_STRIDE + bK0]) = h;
            }
        }
    };

    v8f acc[2][NF];
    #pragma unroll
    for (int mr = 0; mr < 2; ++mr)
        #pragma unroll
        for (int f = 0; f < NF; ++f)
            acc[mr][f] = (v8f)(0.0f);

    loadTile(0);
    storeTile(0);
    __syncthreads();

    int cur = 0;
    for (int k0 = 0; k0 < K; k0 += 32) {
        const bool more = (k0 + 32) < K;     // uniform
        if (more) loadTile(k0 + 32);         // global loads in flight during WMMAs

        // ---- A fragments (16x32 f16): e<8 -> k=e+8*hi ; e>=8 -> k=(e-8)+16+8*hi
        int rowA = wm * 32 + lo;
        const _Float16* pa0 = &As[cur][(rowA)      * LDS_STRIDE + 8 * hi];
        const _Float16* pa1 = &As[cur][(rowA + 16) * LDS_STRIDE + 8 * hi];
        v8h a0l = *(const v8h*)(pa0);
        v8h a0h = *(const v8h*)(pa0 + 16);
        v8h a1l = *(const v8h*)(pa1);
        v8h a1h = *(const v8h*)(pa1 + 16);
        v16h a0 = __builtin_shufflevector(a0l, a0h, 0,1,2,3,4,5,6,7,8,9,10,11,12,13,14,15);
        v16h a1 = __builtin_shufflevector(a1l, a1h, 0,1,2,3,4,5,6,7,8,9,10,11,12,13,14,15);

        // ---- B fragments (32x16 f16): lane=col, e<8 -> k=e+8*hi ; e>=8 -> k=(e-8)+16+8*hi
        v16h bfr[NF];
        #pragma unroll
        for (int f = 0; f < NF; ++f) {
            int colB = wn * (BN / 2) + f * 16 + lo;
            const _Float16* pb = &Bs[cur][colB * LDS_STRIDE + 8 * hi];
            v8h bl = *(const v8h*)(pb);
            v8h bh = *(const v8h*)(pb + 16);
            bfr[f] = __builtin_shufflevector(bl, bh, 0,1,2,3,4,5,6,7,8,9,10,11,12,13,14,15);
        }

        #pragma unroll
        for (int f = 0; f < NF; ++f)
            acc[0][f] = __builtin_amdgcn_wmma_f32_16x16x32_f16(false, a0, false, bfr[f], (short)0, acc[0][f], false, false);
        #pragma unroll
        for (int f = 0; f < NF; ++f)
            acc[1][f] = __builtin_amdgcn_wmma_f32_16x16x32_f16(false, a1, false, bfr[f], (short)0, acc[1][f], false, false);

        if (more) storeTile(cur ^ 1);        // f16 convert + LDS store after WMMAs
        __syncthreads();
        cur ^= 1;
    }

    // ---- epilogue: C/D layout -> row = r + 8*hi, col = lo (per 16x16 frag)
    #pragma unroll
    for (int mr = 0; mr < 2; ++mr) {
        #pragma unroll
        for (int f = 0; f < NF; ++f) {
            int col = tn + wn * (BN / 2) + f * 16 + lo;
            if (col >= N) continue;
            float bv = bias ? bias[col] : 0.0f;
            #pragma unroll
            for (int r = 0; r < 8; ++r) {
                int row = tm + wm * 32 + mr * 16 + r + 8 * hi;
                if (row >= M) continue;
                float v = acc[mr][f][r] * alpha + bv;
                if (gelu) v = 0.5f * v * (1.0f + erff(v * 0.70710678118654752f));
                if (Res)  v += Res[(size_t)row * ldres + col];
                C[(size_t)row * ldc + col] = v;
            }
        }
    }
}

// =====================================================================
// patchify: X (B,224,224) -> patches (B*196, 256)
// =====================================================================
__global__ __launch_bounds__(256) void patchify_kernel(
    const float* __restrict__ X, float* __restrict__ out, int total)
{
    int i = blockIdx.x * 256 + threadIdx.x;
    if (i >= total) return;
    int c  = i & 15;
    int t  = i >> 4;
    int r  = t & 15;  t >>= 4;
    int px = t % NPAT; t /= NPAT;
    int py = t % NPAT;
    int b  = t / NPAT;
    out[i] = X[((size_t)b * IMG + (py * PSZ + r)) * IMG + (px * PSZ + c)];
}

// =====================================================================
// assemble: x[b,s,d] = (s==0 ? cls[d] : tok[b,s-1,d]) + sinusoidal_pos(s,d)
// =====================================================================
__global__ __launch_bounds__(256) void assemble_kernel(
    const float* __restrict__ tok, const float* __restrict__ cls,
    float* __restrict__ x, int total)
{
    int i = blockIdx.x * 256 + threadIdx.x;
    if (i >= total) return;
    int d = i % DMODEL;
    int t = i / DMODEL;
    int s = t % SEQ;
    int b = t / SEQ;
    float base = (s == 0) ? cls[d]
                          : tok[((size_t)b * (SEQ - 1) + (s - 1)) * DMODEL + d];
    float fd   = (float)(d & ~1);
    float freq = __powf(10000.0f, fd * (1.0f / (float)DMODEL));
    float ang  = (float)s / freq;
    float pos  = (d & 1) ? __cosf(ang) : __sinf(ang);
    x[i] = base + pos;
}

// =====================================================================
// LayerNorm: one block (256 thr) per row of length D=768
// =====================================================================
__global__ __launch_bounds__(256) void ln_kernel(
    const float* __restrict__ x, const float* __restrict__ g,
    const float* __restrict__ beta, float* __restrict__ out, int d)
{
    int row = blockIdx.x;
    const float* xr = x + (size_t)row * d;
    float* orow = out + (size_t)row * d;
    __shared__ float red[8];
    __shared__ float stats[2];
    int tid = threadIdx.x, lane = tid & 31, wave = tid >> 5;

    float s = 0.0f;
    for (int j = tid; j < d; j += 256) s += xr[j];
    for (int m = 16; m; m >>= 1) s += __shfl_xor(s, m);
    if (lane == 0) red[wave] = s;
    __syncthreads();
    if (wave == 0) {
        float t = (lane < 8) ? red[lane] : 0.0f;
        for (int m = 4; m; m >>= 1) t += __shfl_xor(t, m);
        if (lane == 0) stats[0] = t / (float)d;
    }
    __syncthreads();
    float mu = stats[0];

    float v = 0.0f;
    for (int j = tid; j < d; j += 256) { float dd = xr[j] - mu; v += dd * dd; }
    for (int m = 16; m; m >>= 1) v += __shfl_xor(v, m);
    if (lane == 0) red[wave] = v;
    __syncthreads();
    if (wave == 0) {
        float t = (lane < 8) ? red[lane] : 0.0f;
        for (int m = 4; m; m >>= 1) t += __shfl_xor(t, m);
        if (lane == 0) stats[1] = t / (float)d;
    }
    __syncthreads();
    float rstd = rsqrtf(stats[1] + 1e-5f);
    for (int j = tid; j < d; j += 256)
        orow[j] = (xr[j] - mu) * rstd * g[j] + beta[j];
}

// =====================================================================
// softmax: one wave (32 lanes) per row; n valid elems, ld row stride
// =====================================================================
__global__ __launch_bounds__(256) void softmax_kernel(
    float* __restrict__ sc, int rows, int n, int ld)
{
    int row = blockIdx.x * 8 + (threadIdx.x >> 5);
    if (row >= rows) return;
    int lane = threadIdx.x & 31;
    float* p = sc + (size_t)row * ld;

    float vals[7];
    float mx = -1e30f;
    int cnt = 0;
    for (int j = lane; j < n; j += 32) { vals[cnt] = p[j]; mx = fmaxf(mx, vals[cnt]); ++cnt; }
    for (int m = 16; m; m >>= 1) mx = fmaxf(mx, __shfl_xor(mx, m));
    float sum = 0.0f;
    cnt = 0;
    for (int j = lane; j < n; j += 32) { float e = __expf(vals[cnt] - mx); vals[cnt] = e; sum += e; ++cnt; }
    for (int m = 16; m; m >>= 1) sum += __shfl_xor(sum, m);
    float inv = 1.0f / sum;
    cnt = 0;
    for (int j = lane; j < n; j += 32) { p[j] = vals[cnt] * inv; ++cnt; }
}

// =====================================================================
// rearrange heads: o[(h,b),s,e] -> oc[(b,s), h*DQ+e]
// =====================================================================
__global__ __launch_bounds__(256) void rearrange_kernel(
    const float* __restrict__ o, float* __restrict__ oc, int total)
{
    int i = blockIdx.x * 256 + threadIdx.x;
    if (i >= total) return;
    int d = i % DMODEL;
    int t = i / DMODEL;
    int s = t % SEQ;
    int b = t / SEQ;
    int h = d >> 6;
    int e = d & 63;
    oc[i] = o[(((size_t)(h * BATCH + b)) * SEQ + s) * DQ + e];
}

// =====================================================================
// host-side helper
// =====================================================================
static inline void launch_gemm(hipStream_t st,
    const float* A, int lda, long long sAz,
    const float* B, int ldb, long long sBz, int transB,
    float* C, int ldc, long long sCz,
    const float* bias, long long sBiasz,
    const float* Res, int ldres,
    int M, int N, int K, int Z, float alpha, int gelu)
{
    dim3 block(256);
    if (N > 64) {
        dim3 grid((N + 127) / 128, (M + 127) / 128, Z);
        gemm_wmma_kernel<128><<<grid, block, 0, st>>>(A, lda, sAz, B, ldb, sBz, transB,
                                                      C, ldc, sCz, bias, sBiasz, Res, ldres,
                                                      M, N, K, alpha, gelu);
    } else {
        dim3 grid((N + 63) / 64, (M + 127) / 128, Z);
        gemm_wmma_kernel<64><<<grid, block, 0, st>>>(A, lda, sAz, B, ldb, sBz, transB,
                                                     C, ldc, sCz, bias, sBiasz, Res, ldres,
                                                     M, N, K, alpha, gelu);
    }
}

extern "C" void kernel_launch(void* const* d_in, const int* in_sizes, int n_in,
                              void* d_out, int out_size, void* d_ws, size_t ws_size,
                              hipStream_t stream)
{
    (void)in_sizes; (void)n_in; (void)out_size; (void)ws_size;
    const float* X       = (const float*)d_in[0];
    const float* patch_W = (const float*)d_in[1];
    const float* patch_b = (const float*)d_in[2];
    const float* cls_tok = (const float*)d_in[3];
    const float* ln1_g   = (const float*)d_in[4];
    const float* ln1_b   = (const float*)d_in[5];
    const float* Wq      = (const float*)d_in[6];
    const float* bq      = (const float*)d_in[7];
    const float* Wk      = (const float*)d_in[8];
    const float* bk      = (const float*)d_in[9];
    const float* Wv      = (const float*)d_in[10];
    const float* bv      = (const float*)d_in[11];
    const float* proj_W  = (const float*)d_in[12];
    const float* proj_b  = (const float*)d_in[13];
    const float* ln2_g   = (const float*)d_in[14];
    const float* ln2_b   = (const float*)d_in[15];
    const float* mlp_W1  = (const float*)d_in[16];
    const float* mlp_b1  = (const float*)d_in[17];
    const float* mlp_W2  = (const float*)d_in[18];
    const float* mlp_b2  = (const float*)d_in[19];
    const float* head_W  = (const float*)d_in[20];
    const float* head_b  = (const float*)d_in[21];
    float* out = (float*)d_out;

    float* ws = (float*)d_ws;
    const size_t NX  = (size_t)BATCH * SEQ * DMODEL;              // 4.84M floats
    const size_t NSC = (size_t)NHEADS * BATCH * SEQ * SEQP;       // padded scores
    float* xb  = ws;            // activations (B*S, D)
    float* nb  = ws + 1 * NX;   // layernorm output
    float* qb  = ws + 2 * NX;   // (NH, B*S, DQ)
    float* kb  = ws + 3 * NX;
    float* vb  = ws + 4 * NX;
    float* ob  = ws + 5 * NX;   // attn output, (NH*B, S, DQ)
    float* ocb = ws + 6 * NX;   // head-concat, (B*S, D)
    float* scb = ws + 7 * NX;   // scores (NH*B, S, SEQP)
    float* hb  = scb + NSC;     // mlp hidden (B*S, MLPH)
    float* patb = scb;          // alias: patches used only before loop
    float* tokb = hb;           // alias: tok used only before loop

    const int BS = BATCH * SEQ;           // 6304
    const int NTOK = BATCH * (SEQ - 1);   // 6272

    // ---- patchify + patch embedding + assemble x with cls & pos-emb
    {
        int total = NTOK * PDIM;
        patchify_kernel<<<(total + 255) / 256, 256, 0, stream>>>(X, patb, total);
    }
    launch_gemm(stream, patb, PDIM, 0, patch_W, DMODEL, 0, 0,
                tokb, DMODEL, 0, patch_b, 0, nullptr, 0,
                NTOK, DMODEL, PDIM, 1, 1.0f, 0);
    {
        int total = BS * DMODEL;
        assemble_kernel<<<(total + 255) / 256, 256, 0, stream>>>(tokb, cls_tok, xb, total);
    }

    const long long sQKV = (long long)BS * DQ;       // C batch stride (per head)
    const long long sAtt = (long long)SEQ * DQ;      // per (h,b) q/k/v/o stride
    const long long sSc  = (long long)SEQ * SEQP;

    for (int l = 0; l < LAYERS; ++l) {
        const float* Wq_l = Wq + (size_t)l * NHEADS * DMODEL * DQ;
        const float* Wk_l = Wk + (size_t)l * NHEADS * DMODEL * DQ;
        const float* Wv_l = Wv + (size_t)l * NHEADS * DMODEL * DQ;
        const float* bq_l = bq + (size_t)l * NHEADS * DQ;
        const float* bk_l = bk + (size_t)l * NHEADS * DQ;
        const float* bv_l = bv + (size_t)l * NHEADS * DQ;

        // LN1
        ln_kernel<<<BS, 256, 0, stream>>>(xb, ln1_g + (size_t)l * DMODEL,
                                          ln1_b + (size_t)l * DMODEL, nb, DMODEL);
        // QKV, batched over heads (A shared, B/bias/C strided per head)
        launch_gemm(stream, nb, DMODEL, 0, Wq_l, DQ, (long long)DMODEL * DQ, 0,
                    qb, DQ, sQKV, bq_l, DQ, nullptr, 0, BS, DQ, DMODEL, NHEADS, 1.0f, 0);
        launch_gemm(stream, nb, DMODEL, 0, Wk_l, DQ, (long long)DMODEL * DQ, 0,
                    kb, DQ, sQKV, bk_l, DQ, nullptr, 0, BS, DQ, DMODEL, NHEADS, 1.0f, 0);
        launch_gemm(stream, nb, DMODEL, 0, Wv_l, DQ, (long long)DMODEL * DQ, 0,
                    vb, DQ, sQKV, bv_l, DQ, nullptr, 0, BS, DQ, DMODEL, NHEADS, 1.0f, 0);
        // scores = (Q K^T) / sqrt(DQ), batched over (h,b) = 384
        launch_gemm(stream, qb, DQ, sAtt, kb, DQ, sAtt, 1,
                    scb, SEQP, sSc, nullptr, 0, nullptr, 0,
                    SEQ, SEQ, DQ, NHEADS * BATCH, 0.125f, 0);
        // softmax over last dim
        {
            int rows = NHEADS * BATCH * SEQ;
            softmax_kernel<<<(rows + 7) / 8, 256, 0, stream>>>(scb, rows, SEQ, SEQP);
        }
        // o = att @ V
        launch_gemm(stream, scb, SEQP, sSc, vb, DQ, sAtt, 0,
                    ob, DQ, sAtt, nullptr, 0, nullptr, 0,
                    SEQ, DQ, SEQ, NHEADS * BATCH, 1.0f, 0);
        // concat heads
        {
            int total = BS * DMODEL;
            rearrange_kernel<<<(total + 255) / 256, 256, 0, stream>>>(ob, ocb, total);
        }
        // x = x + o @ proj_W + proj_b
        launch_gemm(stream, ocb, DMODEL, 0, proj_W + (size_t)l * DMODEL * DMODEL, DMODEL, 0, 0,
                    xb, DMODEL, 0, proj_b + (size_t)l * DMODEL, 0, xb, DMODEL,
                    BS, DMODEL, DMODEL, 1, 1.0f, 0);
        // LN2
        ln_kernel<<<BS, 256, 0, stream>>>(xb, ln2_g + (size_t)l * DMODEL,
                                          ln2_b + (size_t)l * DMODEL, nb, DMODEL);
        // h = gelu(n2 @ W1 + b1)
        launch_gemm(stream, nb, DMODEL, 0, mlp_W1 + (size_t)l * DMODEL * MLPH, MLPH, 0, 0,
                    hb, MLPH, 0, mlp_b1 + (size_t)l * MLPH, 0, nullptr, 0,
                    BS, MLPH, DMODEL, 1, 1.0f, 1);
        // x = x + h @ W2 + b2
        launch_gemm(stream, hb, MLPH, 0, mlp_W2 + (size_t)l * MLPH * DMODEL, DMODEL, 0, 0,
                    xb, DMODEL, 0, mlp_b2 + (size_t)l * DMODEL, 0, xb, DMODEL,
                    BS, DMODEL, MLPH, 1, 1.0f, 0);
    }

    // head: out = x[:,0,:] @ head_W + head_b   (cls rows via lda = S*D)
    launch_gemm(stream, xb, SEQ * DMODEL, 0, head_W, NCLS, 0, 0,
                out, NCLS, 0, head_b, 0, nullptr, 0,
                BATCH, NCLS, DMODEL, 1, 1.0f, 0);
}
